// ViT_33741263077602
// MI455X (gfx1250) — matrix operand order repaired and checked by reading, not compiled, not run on hardware
//
#include <hip/hip_runtime.h>
#include <hip/hip_bf16.h>

typedef unsigned short u16;
typedef __attribute__((ext_vector_type(16))) __bf16 bf16x16;
typedef __attribute__((ext_vector_type(8)))  float  f32x8;
typedef __attribute__((ext_vector_type(4)))  unsigned int uintx4;
typedef __attribute__((ext_vector_type(4)))  unsigned int v4u;
typedef __attribute__((ext_vector_type(8)))  int v8i;
typedef __attribute__((ext_vector_type(4)))  int v4i;

#define B_    32
#define NTOK_ 197
#define NP_   196
#define E_    768
#define H_    3072
#define NH_   12
#define HD_   64
#define KT_   224   // tokens padded to 7*32 for the PV WMMA K-loop

#if defined(__has_builtin)
#if __has_builtin(__builtin_amdgcn_tensor_load_to_lds)
#define USE_TDM 1
#endif
#endif
#ifndef USE_TDM
#define USE_TDM 0
#endif

union Frag { bf16x16 v; u16 u[16]; uintx4 q[2]; };

__device__ inline uintx4 zero4() { uintx4 z = {0u,0u,0u,0u}; return z; }

__device__ inline u16 f2bfu(float f) {
  unsigned u = __builtin_bit_cast(unsigned, f);
  unsigned r = u + 0x7FFFu + ((u >> 16) & 1u);
  return (u16)(r >> 16);
}
__device__ inline float bf2f(u16 h) {
  unsigned u = ((unsigned)h) << 16;
  return __builtin_bit_cast(float, u);
}

__device__ inline f32x8 wmma_bf16(bf16x16 a, bf16x16 b, f32x8 c) {
  // v_wmma_f32_16x16x32_bf16: D = A(16x32) * B(32x16) + C
  return __builtin_amdgcn_wmma_f32_16x16x32_bf16(false, a, false, b, (short)0, c, false, false);
}

// intra-wave LDS write->read ordering (no workgroup barrier needed)
__device__ inline void lds_wave_fence() {
  __builtin_amdgcn_wave_barrier();
  asm volatile("s_wait_dscnt 0" ::: "memory");
  __builtin_amdgcn_wave_barrier();
}

__device__ inline void wait_tensorcnt0() {
#if defined(__has_builtin)
#if __has_builtin(__builtin_amdgcn_s_wait_tensorcnt)
  __builtin_amdgcn_s_wait_tensorcnt(0);
#else
  asm volatile("s_wait_tensorcnt 0" ::: "memory");
#endif
#else
  asm volatile("s_wait_tensorcnt 0" ::: "memory");
#endif
}

#if USE_TDM
// TDM: DMA a 128-row x 64-elem bf16 tile (row stride K elems) from global to LDS.
// D# per CDNA5 ISA ch.8: group0 = {count/flags, lds_addr, global_addr, type=2},
// group1 = {data_size=2B, tensor_dim0=K, tensor_dim1=N, tile 64x128, dim0_stride=K}.
__device__ inline void tdm_load_tile_128x64(u16* lds_dst, const u16* gsrc, int K, int N) {
  unsigned lds_addr = (unsigned)(size_t)lds_dst;
  unsigned long long ga = (unsigned long long)(size_t)gsrc;
  v4u g0;
  g0[0] = 1u;                                              // count=1, no gather
  g0[1] = lds_addr;                                        // lds_addr [63:32]
  g0[2] = (unsigned)ga;                                    // global_addr lo
  g0[3] = ((unsigned)(ga >> 32) & 0x01FFFFFFu) | (2u << 30); // ga hi | type=2
  unsigned uK = (unsigned)K, uN = (unsigned)N;
  v8i g1;
  g1[0] = (int)0x00010000u;                                // data_size=1 (2 bytes)
  g1[1] = (int)((uK & 0xFFFFu) << 16);                     // tensor_dim0[15:0] @ [63:48]
  g1[2] = (int)((uK >> 16) | ((uN & 0xFFFFu) << 16));      // dim0[31:16], dim1[15:0]
  g1[3] = (int)((uN >> 16) | (64u << 16));                 // dim1[31:16], tile_dim0=64
  g1[4] = (int)128u;                                       // tile_dim1=128, tile_dim2=0
  g1[5] = (int)uK;                                         // tensor_dim0_stride lo
  g1[6] = 0;                                               // stride hi, dim1_stride lo
  g1[7] = 0;
  v4i z4 = {0, 0, 0, 0};
#if __clang_major__ >= 23
  v8i z8 = {0, 0, 0, 0, 0, 0, 0, 0};
  __builtin_amdgcn_tensor_load_to_lds(g0, g1, z4, z4, z8, 0);
#else
  __builtin_amdgcn_tensor_load_to_lds(g0, g1, z4, z4, 0);
#endif
}
#endif

// ---------------------------------------------------------------------------
// fp32 (KxN) -> bf16 transposed (NxK), tiled 32x32 through LDS.
__global__ __launch_bounds__(256) void cvtT_kernel(const float* __restrict__ in,
                                                   u16* __restrict__ out, int K, int N) {
  __shared__ float tile[32][33];
  int nb = blockIdx.x * 32, kb = blockIdx.y * 32;
  int tx = threadIdx.x, ty = threadIdx.y;     // 32 x 8
  for (int i = ty; i < 32; i += 8)
    tile[i][tx] = in[(size_t)(kb + i) * N + nb + tx];
  __syncthreads();
  for (int i = ty; i < 32; i += 8)
    out[(size_t)(nb + i) * K + kb + tx] = f2bfu(tile[tx][i]);
}

// ---------------------------------------------------------------------------
// im2col: (B,224,224,3) -> (B*196, 768) bf16, col = (py*16+px)*3+c
__global__ void im2col_kernel(const float* __restrict__ x, u16* __restrict__ out, int total) {
  int idx = blockIdx.x * blockDim.x + threadIdx.x;
  if (idx >= total) return;
  int col = idx % E_; int rest = idx / E_;
  int p = rest % NP_; int b = rest / NP_;
  int c = col % 3; int pxy = col / 3;
  int px = pxy % 16, py = pxy / 16;
  int gx = p % 14, gy = p / 14;
  int row = gy * 16 + py, cc = gx * 16 + px;
  out[idx] = f2bfu(x[(((size_t)b * 224 + row) * 224 + cc) * 3 + c]);
}

// ---------------------------------------------------------------------------
// h = concat(cls, patch_out) + pos_embed
__global__ void assemble_kernel(const float* __restrict__ tmp, const float* __restrict__ cls,
                                const float* __restrict__ pos, float* __restrict__ h, int total) {
  int idx = blockIdx.x * blockDim.x + threadIdx.x;
  if (idx >= total) return;
  int e = idx % E_; int rest = idx / E_;
  int t = rest % NTOK_; int b = rest / NTOK_;
  float v = (t == 0) ? cls[e] : tmp[((size_t)b * NP_ + (t - 1)) * E_ + e];
  h[idx] = v + pos[t * E_ + e];
}

// ---------------------------------------------------------------------------
// bf16 GEMM with PRE-TRANSPOSED weights:
//   C(MxN,f32) = A(MxK,bf16) @ Bt(NxK,bf16)^T + bias [+ residual]
// 256 threads / 8 waves; 128x128 tile; BK=64; wave tile 32x64.
// Bt tile staged by the Tensor Data Mover (wave 0 issues, TENSORcnt-waited);
// A tile staged per-thread (ragged-M guard). 16 WMMAs per barrier pair.
#define BM 128
#define BN 128
#define BK 64
__global__ __launch_bounds__(256) void gemm_bf16_kernel(
    const u16* __restrict__ A, const u16* __restrict__ Bt,
    const float* __restrict__ bias, const float* __restrict__ residual,
    float* __restrict__ Cf, u16* __restrict__ Cb, u16* __restrict__ CvT,
    int M, int N, int K) {
  __shared__ u16 sA[BM * BK];   // row-major, stride BK
  __shared__ u16 sB[BN * BK];   // Bt tile: sB[n*BK+k] (K-contiguous rows)
  int tid = threadIdx.x;
  int wave = tid >> 5, lane = tid & 31;
  int lhalf = lane >> 4, l15 = lane & 15;
  int m0 = blockIdx.y * BM, n0 = blockIdx.x * BN;
  int wm = (wave >> 1) * 32, wn = (wave & 1) * 64;
  f32x8 acc[2][4] = {};
  int ldr = tid >> 1;                // 0..127
  int ldc = (tid & 1) * 32;          // 0 or 32
  for (int k0 = 0; k0 < K; k0 += BK) {
#if USE_TDM
    if (wave == 0)  // TDM ignores EXEC; single-wave branch => one DMA per tile
      tdm_load_tile_128x64(sB, Bt + (size_t)n0 * K + k0, K, N);
#else
    { // per-thread Bt staging fallback
      const u16* src = Bt + (size_t)(n0 + ldr) * K + k0 + ldc;
      u16* d = &sB[ldr * BK + ldc];
      *(uintx4*)(d) = *(const uintx4*)src;
      *(uintx4*)(d + 8) = *(const uintx4*)(src + 8);
      *(uintx4*)(d + 16) = *(const uintx4*)(src + 16);
      *(uintx4*)(d + 24) = *(const uintx4*)(src + 24);
    }
#endif
    { // stage A tile (guard ragged M)
      uintx4 a0 = zero4(), a1 = zero4(), a2 = zero4(), a3 = zero4();
      if (m0 + ldr < M) {
        const u16* src = A + (size_t)(m0 + ldr) * K + k0 + ldc;
        a0 = *(const uintx4*)src;        a1 = *(const uintx4*)(src + 8);
        a2 = *(const uintx4*)(src + 16); a3 = *(const uintx4*)(src + 24);
        if (k0 + BK < K) __builtin_prefetch(src + BK, 0, 3);
      }
      u16* d = &sA[ldr * BK + ldc];
      *(uintx4*)(d) = a0; *(uintx4*)(d + 8) = a1;
      *(uintx4*)(d + 16) = a2; *(uintx4*)(d + 24) = a3;
    }
#if USE_TDM
    if (wave == 0) wait_tensorcnt0();
#endif
    __syncthreads();
#pragma unroll
    for (int ks = 0; ks < 2; ks++) {
      // load ALL fragments into distinct registers, then issue 8 WMMAs:
      // gives the scheduler room to hide LDS latency instead of wait-0 chains.
      Frag af[2], bf[4];
#pragma unroll
      for (int i = 0; i < 2; i++) {
        int arow = wm + i * 16 + l15;
        int ak = ks * 32 + lhalf * 8;
        af[i].q[0] = *(const uintx4*)(&sA[arow * BK + ak]);
        af[i].q[1] = *(const uintx4*)(&sA[arow * BK + ak + 16]);
      }
#pragma unroll
      for (int j = 0; j < 4; j++) {
        int brow = wn + j * 16 + l15;
        int bk = ks * 32 + lhalf * 8;
        bf[j].q[0] = *(const uintx4*)(&sB[brow * BK + bk]);
        bf[j].q[1] = *(const uintx4*)(&sB[brow * BK + bk + 16]);
      }
#pragma unroll
      for (int j = 0; j < 4; j++) {
        acc[0][j] = wmma_bf16(af[0].v, bf[j].v, acc[0][j]);
        acc[1][j] = wmma_bf16(af[1].v, bf[j].v, acc[1][j]);
      }
    }
    __syncthreads();
  }
  // epilogue: C-layout row = base + lhalf*8 + r, col = base + (lane&15)
#pragma unroll
  for (int i = 0; i < 2; i++)
#pragma unroll
    for (int j = 0; j < 4; j++) {
      int col = n0 + wn + j * 16 + l15;
      float bv = bias ? bias[col] : 0.f;
#pragma unroll
      for (int r = 0; r < 8; r++) {
        int row = m0 + wm + i * 16 + lhalf * 8 + r;
        if (row < M) {
          size_t idx = (size_t)row * N + col;
          float v = acc[i][j][r] + bv;
          if (residual) v += residual[idx];
          if (Cf) Cf[idx] = v;
          if (Cb) Cb[idx] = f2bfu(v);
          if (CvT) {  // V in (b, h, d, t) layout for attention PV matmul
            int bb = row / NTOK_, tt = row - bb * NTOK_;
            int hh = col >> 6, dd = col & 63;
            CvT[(((size_t)bb * NH_ + hh) * HD_ + dd) * KT_ + tt] = f2bfu(v);
          }
        }
      }
    }
}

// ---------------------------------------------------------------------------
// LayerNorm (fp32 in -> bf16 out), one block per row
__global__ __launch_bounds__(256) void ln_kernel(
    const float* __restrict__ x, const float* __restrict__ s, const float* __restrict__ b,
    u16* __restrict__ out, int width, float eps) {
  __shared__ float red[66];
  int row = blockIdx.x;
  const float* xr = x + (size_t)row * width;
  float sum = 0.f, sum2 = 0.f;
  for (int i = threadIdx.x; i < width; i += blockDim.x) {
    float v = xr[i]; sum += v; sum2 += v * v;
  }
#pragma unroll
  for (int o = 16; o; o >>= 1) { sum += __shfl_xor(sum, o, 32); sum2 += __shfl_xor(sum2, o, 32); }
  int wave = threadIdx.x >> 5, lane = threadIdx.x & 31;
  if (lane == 0) { red[wave] = sum; red[wave + 32] = sum2; }
  __syncthreads();
  if (threadIdx.x == 0) {
    float a = 0.f, c = 0.f;
    int nw = blockDim.x >> 5;
    for (int i = 0; i < nw; i++) { a += red[i]; c += red[i + 32]; }
    float mean = a / width;
    float var = c / width - mean * mean;
    red[64] = mean; red[65] = rsqrtf(var + eps);
  }
  __syncthreads();
  float mean = red[64], inv = red[65];
  for (int i = threadIdx.x; i < width; i += blockDim.x)
    out[(size_t)row * width + i] = f2bfu((xr[i] - mean) * inv * s[i] + b[i]);
}

// ---------------------------------------------------------------------------
// 2D RoPE on patch tokens (cls passthrough); fp32 Q/K -> bf16 q/k
__global__ void rope_kernel(const float* __restrict__ Qf, const float* __restrict__ Kf,
                            u16* __restrict__ qb, u16* __restrict__ kb, int total) {
  int idx = blockIdx.x * blockDim.x + threadIdx.x;
  if (idx >= total) return;          // total = B*NTOK*NH*32 pairs
  int p = idx & 31; int rest = idx >> 5;
  int hh = rest % NH_; rest /= NH_;
  int t = rest % NTOK_; int b = rest / NTOK_;
  size_t off = ((size_t)(b * NTOK_ + t)) * E_ + hh * HD_ + 2 * p;
  float q0 = Qf[off], q1 = Qf[off + 1];
  float k0 = Kf[off], k1 = Kf[off + 1];
  if (t == 0) {
    qb[off] = f2bfu(q0); qb[off + 1] = f2bfu(q1);
    kb[off] = f2bfu(k0); kb[off + 1] = f2bfu(k1);
    return;
  }
  int pi = t - 1, gi = pi / 14, gj = pi % 14;
  int d0 = 2 * p;
  int half = d0 >> 5;
  int mf = (d0 & 31) >> 1;
  float freq = __powf(10000.f, -(float)(2 * mf) / 32.f);
  float tt = half ? (float)gj : (float)gi;
  float ang = tt * freq;
  float cs = __cosf(ang), sn = __sinf(ang);
  qb[off]     = f2bfu(q0 * cs - q1 * sn);
  qb[off + 1] = f2bfu(q1 * cs + q0 * sn);
  kb[off]     = f2bfu(k0 * cs - k1 * sn);
  kb[off + 1] = f2bfu(k1 * cs + k0 * sn);
}

// ---------------------------------------------------------------------------
// Flash-style attention, one workgroup per (b, head), 8 waves, all-WMMA.
__global__ __launch_bounds__(256) void attn_kernel(
    const u16* __restrict__ qb, const u16* __restrict__ kb,
    const u16* __restrict__ vtg, u16* __restrict__ ob) {
  __shared__ u16 sP[8][16 * 32];      // per-wave P tile bounce, 8KB
  int bh = blockIdx.x;
  int b = bh / NH_, hh = bh % NH_;
  int tid = threadIdx.x, wave = tid >> 5, lane = tid & 31;
  int lhalf = lane >> 4, l15 = lane & 15;
  size_t base = ((size_t)b * NTOK_) * E_ + hh * HD_;
  const u16* vt = vtg + (size_t)bh * HD_ * KT_;
  const float scale = 0.125f;  // 64^-0.5
  for (int qt = wave; qt < 13; qt += 8) {
    Frag aq[2];
    int qrow = qt * 16 + l15;
#pragma unroll
    for (int s = 0; s < 2; s++) {
      if (qrow < NTOK_) {
        const u16* src = qb + base + (size_t)qrow * E_ + s * 32 + lhalf * 8;
        aq[s].q[0] = *(const uintx4*)src;
        aq[s].q[1] = *(const uintx4*)(src + 16);
      } else { aq[s].q[0] = zero4(); aq[s].q[1] = zero4(); }
    }
    f32x8 accO[4] = {};
    float m[8], l[8];
#pragma unroll
    for (int r = 0; r < 8; r++) { m[r] = -1e30f; l[r] = 0.f; }
    for (int jp = 0; jp < 7; jp++) {           // 7 * 32 = 224 padded keys
      f32x8 s0 = {}, s1 = {};
      int j0 = jp * 32 + l15, j1 = j0 + 16;
#pragma unroll
      for (int s = 0; s < 2; s++) {            // head-dim in two K=32 steps
        Frag b0, b1;
        if (j0 < NTOK_) {
          const u16* p = kb + base + (size_t)j0 * E_ + s * 32 + lhalf * 8;
          b0.q[0] = *(const uintx4*)p; b0.q[1] = *(const uintx4*)(p + 16);
        } else { b0.q[0] = zero4(); b0.q[1] = zero4(); }
        if (j1 < NTOK_) {
          const u16* p = kb + base + (size_t)j1 * E_ + s * 32 + lhalf * 8;
          b1.q[0] = *(const uintx4*)p; b1.q[1] = *(const uintx4*)(p + 16);
        } else { b1.q[0] = zero4(); b1.q[1] = zero4(); }
        s0 = wmma_bf16(aq[s].v, b0.v, s0);
        s1 = wmma_bf16(aq[s].v, b1.v, s1);
      }
      bool v0 = j0 < NTOK_, v1 = j1 < NTOK_;
#pragma unroll
      for (int r = 0; r < 8; r++) {
        float a0 = v0 ? s0[r] * scale : -1e30f;
        float a1 = v1 ? s1[r] * scale : -1e30f;
        float tm = fmaxf(a0, a1);
        tm = fmaxf(tm, __shfl_xor(tm, 1, 32));
        tm = fmaxf(tm, __shfl_xor(tm, 2, 32));
        tm = fmaxf(tm, __shfl_xor(tm, 4, 32));
        tm = fmaxf(tm, __shfl_xor(tm, 8, 32));   // reduce within 16-lane half
        float nm = fmaxf(m[r], tm);
        float alpha = __expf(m[r] - nm);
        float p0 = __expf(a0 - nm), p1 = __expf(a1 - nm);
        float ps = p0 + p1;
        ps += __shfl_xor(ps, 1, 32); ps += __shfl_xor(ps, 2, 32);
        ps += __shfl_xor(ps, 4, 32); ps += __shfl_xor(ps, 8, 32);
        l[r] = l[r] * alpha + ps;
        m[r] = nm;
#pragma unroll
        for (int dt = 0; dt < 4; dt++) accO[dt][r] *= alpha;
        int prow = lhalf * 8 + r;
        sP[wave][prow * 32 + l15] = f2bfu(p0);
        sP[wave][prow * 32 + 16 + l15] = f2bfu(p1);
      }
      lds_wave_fence();
      Frag ap;  // re-read P as an A fragment (16x32)
      {
        const u16* p = &sP[wave][l15 * 32 + lhalf * 8];
        ap.q[0] = *(const uintx4*)p;
        ap.q[1] = *(const uintx4*)(p + 16);
      }
#pragma unroll
      for (int dt = 0; dt < 4; dt++) {
        Frag bv;   // padded-token slots multiply against P==0, so pad garbage is harmless
        const u16* p = vt + (size_t)(dt * 16 + l15) * KT_ + jp * 32 + lhalf * 8;
        bv.q[0] = *(const uintx4*)p; bv.q[1] = *(const uintx4*)(p + 16);
        accO[dt] = wmma_bf16(ap.v, bv.v, accO[dt]);
      }
      lds_wave_fence();
    }
#pragma unroll
    for (int dt = 0; dt < 4; dt++)
#pragma unroll
      for (int r = 0; r < 8; r++) {
        int tok = qt * 16 + lhalf * 8 + r;
        if (tok < NTOK_) {
          float inv = 1.f / l[r];
          ob[base + (size_t)tok * E_ + dt * 16 + l15] = f2bfu(accO[dt][r] * inv);
        }
      }
  }
}

// ---------------------------------------------------------------------------
// Fused SwiGLU gate + LayerNorm(eps=1e-5): g = LN(silu(u)*g2) -> bf16
__global__ __launch_bounds__(256) void glu_ln_kernel(
    const u16* __restrict__ u, const u16* __restrict__ g2,
    const float* __restrict__ sc, const float* __restrict__ bi,
    u16* __restrict__ out) {
  __shared__ float gv[H_];
  __shared__ float red[66];
  int row = blockIdx.x;
  size_t base = (size_t)row * H_;
  float sum = 0.f, sum2 = 0.f;
  for (int i = threadIdx.x; i < H_; i += 256) {
    float uu = bf2f(u[base + i]);
    float gg = bf2f(g2[base + i]);
    float val = (uu / (1.f + __expf(-uu))) * gg;
    gv[i] = val; sum += val; sum2 += val * val;
  }
#pragma unroll
  for (int o = 16; o; o >>= 1) { sum += __shfl_xor(sum, o, 32); sum2 += __shfl_xor(sum2, o, 32); }
  int wave = threadIdx.x >> 5, lane = threadIdx.x & 31;
  if (lane == 0) { red[wave] = sum; red[wave + 32] = sum2; }
  __syncthreads();
  if (threadIdx.x == 0) {
    float a = 0.f, c = 0.f;
    for (int i = 0; i < 8; i++) { a += red[i]; c += red[i + 32]; }
    float mean = a / H_;
    float var = c / H_ - mean * mean;
    red[64] = mean; red[65] = rsqrtf(var + 1e-5f);
  }
  __syncthreads();
  float mean = red[64], inv = red[65];
  for (int i = threadIdx.x; i < H_; i += 256)
    out[base + i] = f2bfu((gv[i] - mean) * inv * sc[i] + bi[i]);
}

// ===========================================================================
extern "C" void kernel_launch(void* const* d_in, const int* in_sizes, int n_in,
                              void* d_out, int out_size, void* d_ws, size_t ws_size,
                              hipStream_t stream) {
  (void)in_sizes; (void)n_in; (void)out_size; (void)ws_size;
  const float* x        = (const float*)d_in[0];
  const float* patch_w  = (const float*)d_in[1];
  const float* patch_b  = (const float*)d_in[2];
  const float* cls_tok  = (const float*)d_in[3];
  const float* pos_emb  = (const float*)d_in[4];
  const float* q_w      = (const float*)d_in[5];
  const float* k_w      = (const float*)d_in[6];
  const float* v_w      = (const float*)d_in[7];
  const float* q_b      = (const float*)d_in[8];
  const float* v_b      = (const float*)d_in[9];
  const float* proj_w   = (const float*)d_in[10];
  const float* proj_b   = (const float*)d_in[11];
  const float* norm1_s  = (const float*)d_in[12];
  const float* norm1_b  = (const float*)d_in[13];
  const float* norm2_s  = (const float*)d_in[14];
  const float* norm2_b  = (const float*)d_in[15];
  const float* w1_w     = (const float*)d_in[16];
  const float* w1_b     = (const float*)d_in[17];
  const float* w2_w     = (const float*)d_in[18];
  const float* w2_b     = (const float*)d_in[19];
  const float* ffn_s    = (const float*)d_in[20];
  const float* ffn_b    = (const float*)d_in[21];
  const float* w3_w     = (const float*)d_in[22];
  const float* w3_b     = (const float*)d_in[23];
  float* out = (float*)d_out;

  const int M1 = B_ * NTOK_;         // 6304
  const int Mp = B_ * NP_;           // 6272
  const long nEE = (long)E_ * E_;    // 589824
  const long nEH = (long)E_ * H_;    // 2359296

  char* ws = (char*)d_ws;
  size_t off = 0;
  auto alloc = [&](size_t bytes) -> void* {
    void* p = ws + off;
    off += (bytes + 255) & ~(size_t)255;
    return p;
  };
  u16*   pw_bf  = (u16*)  alloc((size_t)nEE * 2);          // patch_w^T
  u16*   Apatch = (u16*)  alloc((size_t)Mp * E_ * 2);
  float* tmp    = (float*)alloc((size_t)Mp * E_ * 4);
  float* h      = (float*)alloc((size_t)M1 * E_ * 4);
  u16*   ybf    = (u16*)  alloc((size_t)M1 * E_ * 2);
  float* Qf     = (float*)alloc((size_t)M1 * E_ * 4);
  float* Kf     = (float*)alloc((size_t)M1 * E_ * 4);
  u16*   qbf    = (u16*)  alloc((size_t)M1 * E_ * 2);
  u16*   kbf    = (u16*)  alloc((size_t)M1 * E_ * 2);
  u16*   vtg    = (u16*)  alloc((size_t)B_ * NH_ * HD_ * KT_ * 2);  // V^T, (b,h,d,t)
  u16*   obf    = (u16*)  alloc((size_t)M1 * E_ * 2);
  u16*   ubf    = (u16*)  alloc((size_t)M1 * H_ * 2);
  u16*   g2bf   = (u16*)  alloc((size_t)M1 * H_ * 2);
  u16*   glnbf  = (u16*)  alloc((size_t)M1 * H_ * 2);
  u16*   wq     = (u16*)  alloc((size_t)nEE * 2);          // all transposed (NxK)
  u16*   wk     = (u16*)  alloc((size_t)nEE * 2);
  u16*   wv     = (u16*)  alloc((size_t)nEE * 2);
  u16*   wp     = (u16*)  alloc((size_t)nEE * 2);
  u16*   w1b    = (u16*)  alloc((size_t)nEH * 2);
  u16*   w2b    = (u16*)  alloc((size_t)nEH * 2);
  u16*   w3b    = (u16*)  alloc((size_t)nEH * 2);

  auto cvtT = [&](const float* src, u16* dst, int K, int N) {
    cvtT_kernel<<<dim3((unsigned)(N / 32), (unsigned)(K / 32)), dim3(32, 8), 0, stream>>>(src, dst, K, N);
  };
  auto gemm = [&](const u16* A, const u16* Bt, const float* bias, const float* res,
                  float* Cf, u16* Cb, u16* CvT, int M, int N, int K) {
    dim3 g((unsigned)(N / 128), (unsigned)((M + 127) / 128));
    gemm_bf16_kernel<<<g, 256, 0, stream>>>(A, Bt, bias, res, Cf, Cb, CvT, M, N, K);
  };

  // ---- patch embed ----
  cvtT(patch_w, pw_bf, E_, E_);
  {
    int tot = Mp * E_;
    im2col_kernel<<<dim3((tot + 255) / 256), 256, 0, stream>>>(x, Apatch, tot);
  }
  gemm(Apatch, pw_bf, patch_b, nullptr, tmp, nullptr, nullptr, Mp, E_, E_);
  {
    int tot = M1 * E_;
    assemble_kernel<<<dim3((tot + 255) / 256), 256, 0, stream>>>(tmp, cls_tok, pos_emb, h, tot);
  }

  // ---- transformer layers ----
  const int rope_total = B_ * NTOK_ * NH_ * 32;
  for (int i = 0; i < 12; i++) {
    cvtT(q_w    + (size_t)i * nEE, wq,  E_, E_);
    cvtT(k_w    + (size_t)i * nEE, wk,  E_, E_);
    cvtT(v_w    + (size_t)i * nEE, wv,  E_, E_);
    cvtT(proj_w + (size_t)i * nEE, wp,  E_, E_);
    cvtT(w1_w   + (size_t)i * nEH, w1b, E_, H_);
    cvtT(w2_w   + (size_t)i * nEH, w2b, E_, H_);
    cvtT(w3_w   + (size_t)i * nEH, w3b, H_, E_);

    // LN1 -> y
    ln_kernel<<<dim3(M1), 256, 0, stream>>>(h, norm1_s + (size_t)i * E_, norm1_b + (size_t)i * E_,
                                            ybf, E_, 1e-6f);
    // QKV
    gemm(ybf, wq, q_b + (size_t)i * E_, nullptr, Qf, nullptr, nullptr, M1, E_, E_);
    gemm(ybf, wk, nullptr,              nullptr, Kf, nullptr, nullptr, M1, E_, E_);
    gemm(ybf, wv, v_b + (size_t)i * E_, nullptr, nullptr, nullptr, vtg, M1, E_, E_);
    // RoPE -> bf16 q/k
    rope_kernel<<<dim3((rope_total + 255) / 256), 256, 0, stream>>>(Qf, Kf, qbf, kbf, rope_total);
    // attention
    attn_kernel<<<dim3(B_ * NH_), 256, 0, stream>>>(qbf, kbf, vtg, obf);
    // proj + residual
    gemm(obf, wp, proj_b + (size_t)i * E_, h, h, nullptr, nullptr, M1, E_, E_);
    // LN2 -> y
    ln_kernel<<<dim3(M1), 256, 0, stream>>>(h, norm2_s + (size_t)i * E_, norm2_b + (size_t)i * E_,
                                            ybf, E_, 1e-6f);
    // FFN up projections
    gemm(ybf, w1b, w1_b + (size_t)i * H_, nullptr, nullptr, ubf,  nullptr, M1, H_, E_);
    gemm(ybf, w2b, w2_b + (size_t)i * H_, nullptr, nullptr, g2bf, nullptr, M1, H_, E_);
    // fused silu-gate + LN(1e-5)
    glu_ln_kernel<<<dim3(M1), 256, 0, stream>>>(ubf, g2bf, ffn_s + (size_t)i * H_,
                                                ffn_b + (size_t)i * H_, glnbf);
    // down projection + residual (last layer writes straight to d_out)
    float* dst = (i == 11) ? out : h;
    gemm(glnbf, w3b, w3_b + (size_t)i * E_, h, dst, nullptr, nullptr, M1, E_, H_);
  }
}
